// FlowEncoder_61272003444860
// MI455X (gfx1250) — matrix operand order repaired and checked by reading, not compiled, not run on hardware
//
#include <hip/hip_runtime.h>
#include <hip/hip_bf16.h>

typedef __attribute__((ext_vector_type(16))) _Float16 v16h;
typedef __attribute__((ext_vector_type(8)))  _Float16 v8h;
typedef __attribute__((ext_vector_type(4)))  _Float16 v4h;
typedef __attribute__((ext_vector_type(8)))  float    v8f;

// ---------------------------------------------------------------------------
// Fold BN into conv weights AND pre-shuffle into WMMA A-fragment lane order.
// Output layout: wf[((mtile*NKS + kstep)*32 + lane)*16 + j]
//   lane(l16) holds row m = mtile*16 + l16
//   element j maps to k_local = (lane>>4)*16 + j   (contiguous, matches B gather)
//   k = kstep*32 + k_local, decomposed as k = kh*KROW + (kw<<L2C) + cin
// ---------------------------------------------------------------------------
__global__ void fold_shuffle_kernel(const float* __restrict__ w, const float* __restrict__ b,
                                    const float* __restrict__ g, const float* __restrict__ be,
                                    const float* __restrict__ m, const float* __restrict__ v,
                                    _Float16* __restrict__ wf, float* __restrict__ bout,
                                    int Cout, int Cin, int KH, int KW, int L2C,
                                    int KROW, int NKS) {
    int i = blockIdx.x * blockDim.x + threadIdx.x;
    if (i < Cout) {
        float s = g[i] * rsqrtf(v[i] + 1e-5f);
        bout[i] = (b[i] - m[i]) * s + be[i];
    }
    int total = (Cout / 16) * NKS * 512;
    if (i >= total) return;
    int mt    = i / (NKS * 512);
    int r     = i - mt * (NKS * 512);
    int kstep = r >> 9;
    int e     = r & 511;
    int lane  = e >> 4;
    int j     = e & 15;
    int half  = lane >> 4;
    int l16   = lane & 15;
    int mrow  = mt * 16 + l16;
    int klocal = half * 16 + j;
    int k   = kstep * 32 + klocal;
    int kh  = k / KROW;
    int rem = k - kh * KROW;
    int kw  = rem >> L2C;
    int cin = rem & ((1 << L2C) - 1);
    float val = 0.f;
    if (kh < KH && kw < KW && cin < Cin) {
        float s = g[mrow] * rsqrtf(v[mrow] + 1e-5f);
        val = w[(((long)mrow * Cin + cin) * KH + kh) * KW + kw] * s;
    }
    wf[i] = (_Float16)val;
}

// ---------------------------------------------------------------------------
// img (B,3,H,W) f32 NCHW -> (B,H,W,4) f16 NHWC; one thread per pixel, b64 store
// ---------------------------------------------------------------------------
__global__ void img_to_nhwc4_kernel(const float* __restrict__ img, _Float16* __restrict__ out,
                                    int Bn, int H, int W) {
    long pix = blockIdx.x * (long)blockDim.x + threadIdx.x;
    long total = (long)Bn * H * W;
    if (pix >= total) return;
    int b = (int)(pix / ((long)H * W));
    int r = (int)(pix - (long)b * H * W);
    const float* p = img + (long)b * 3 * H * W + r;
    v4h o;
    o[0] = (_Float16)p[0];
    o[1] = (_Float16)p[(long)H * W];
    o[2] = (_Float16)p[2 * (long)H * W];
    o[3] = (_Float16)0.f;
    *(v4h*)(out + pix * 4) = o;
}

// ---------------------------------------------------------------------------
// Implicit-GEMM conv + bias + ReLU with WMMA f32_16x16x32_f16, NHWC layouts.
//   grid.x: pixel tiles of 128 (8 waves x 16), grid.y: COUT/(16*MS)
//   Each wave: 1 B fragment gather reused by MS WMMAs (MS M-subtiles).
// ---------------------------------------------------------------------------
template<int CINP, int L2C, int H, int W, int COUT, int HOUT, int WOUT,
         int KH, int KW, int STRIDE, int PAD, int MS>
__global__ __launch_bounds__(256) void conv_wmma_kernel(
        const _Float16* __restrict__ in, const _Float16* __restrict__ wt,
        const float* __restrict__ bias, _Float16* __restrict__ out, int Bn) {
    constexpr int KROW = (KW * CINP + 31) & ~31;     // K per kh row, mult of 32
    constexpr int NKS  = (KH * KROW) / 32;           // total K-steps
    constexpr int HWo  = HOUT * WOUT;

    const int tid  = threadIdx.x;
    const int wave = tid >> 5;
    const int lane = tid & 31;
    const int half = lane >> 4;
    const int l16  = lane & 15;

    const int Npix   = Bn * HWo;
    const int mtile0 = blockIdx.y * MS;
    const int n      = blockIdx.x * 128 + wave * 16 + l16;
    const bool vn    = (n < Npix);
    const int  nn    = vn ? n : 0;

    const int b_idx = nn / HWo;                      // compile-time divisor
    const int r2    = nn - b_idx * HWo;
    const int oh    = r2 / WOUT;
    const int ow    = r2 - oh * WOUT;
    const int ih0   = oh * STRIDE - PAD;
    const int iw0   = ow * STRIDE - PAD;

    // valid rem window for this pixel's row gather (kw bounds folded into rem)
    const int kwmin  = iw0 < 0 ? -iw0 : 0;
    const int kwmax1 = (W - iw0) < KW ? (W - iw0) : KW;
    const int remlo  = kwmin  << L2C;
    const int remhi  = kwmax1 << L2C;

    v8f acc[MS] = {};

    for (int kh = 0; kh < KH; ++kh) {
        const int  ih    = ih0 + kh;
        const bool rowok = vn && ((unsigned)ih < (unsigned)H);
        const long rowoff = (((long)b_idx * H + ih) * W + iw0) * (long)CINP;

        for (int ks = 0; ks < KROW; ks += 32) {
            const int kstep    = kh * (KROW / 32) + (ks >> 5);
            const int rem_base = ks + half * 16;

            // ---- B fragment: 16 contiguous f16 from the NHWC row ----
            v16h bfrag;
            const bool fast = (CINP >= 16) && rowok &&
                              (rem_base >= remlo) && (rem_base + 16 <= remhi);
            if (fast) {
                bfrag = *(const v16h*)(in + rowoff + rem_base);
            } else {
                #pragma unroll
                for (int j = 0; j < 16; ++j) {
                    int rem = rem_base + j;
                    _Float16 val = (_Float16)0.f;
                    if (rowok && rem >= remlo && rem < remhi)
                        val = in[rowoff + rem];
                    bfrag[j] = val;
                }
            }

            // ---- MS A fragments (pre-shuffled, contiguous 32B) + WMMA ----
            #pragma unroll
            for (int s = 0; s < MS; ++s) {
                const _Float16* ap = wt + (((long)(mtile0 + s) * NKS + kstep) << 9)
                                        + (lane << 4);
                v16h afrag = *(const v16h*)ap;
                acc[s] = __builtin_amdgcn_wmma_f32_16x16x32_f16(
                    false, afrag, false, bfrag, (short)0, acc[s], false, false);
            }
            // prefetch next weight fragment tile (global_prefetch_b8)
            __builtin_prefetch(wt + (((long)mtile0 * NKS + kstep + 1) << 9) + (lane << 4), 0, 0);
        }
    }

    if (vn) {
        const long obase = (long)nn * COUT;
        #pragma unroll
        for (int s = 0; s < MS; ++s) {
            const int mb = (mtile0 + s) * 16 + half * 8;   // D: VGPR r -> M = r + 8*half
            v8h o;
            #pragma unroll
            for (int r = 0; r < 8; ++r) {
                float val = acc[s][r] + bias[mb + r];
                o[r] = (_Float16)(val > 0.f ? val : 0.f);
            }
            *(v8h*)(out + obase + mb) = o;
        }
    }
}

// ---------------------------------------------------------------------------
// L2-normalize f0 (B,48,64,128) into zero-padded (B,56,72,128), NHWC, vectorized
// ---------------------------------------------------------------------------
__global__ void l2norm_pad_kernel(const _Float16* __restrict__ f, _Float16* __restrict__ fp,
                                  int Bn) {
    const int H = 48, W = 64, C = 128, PD = 4, Hp = 56, Wp = 72;
    long i = blockIdx.x * (long)blockDim.x + threadIdx.x;
    long total = (long)Bn * Hp * Wp;
    if (i >= total) return;
    int b  = (int)(i / (Hp * Wp));
    int r  = (int)(i - (long)b * Hp * Wp);
    int hp = r / Wp, wp = r - (r / Wp) * Wp;
    int h = hp - PD, w = wp - PD;
    bool inside = (unsigned)h < (unsigned)H && (unsigned)w < (unsigned)W;
    v8h* dst = (v8h*)(fp + i * C);
    if (inside) {
        const v8h* src = (const v8h*)(f + (((long)b * H + h) * W + w) * C);
        float ss = 0.f;
        #pragma unroll
        for (int c8 = 0; c8 < C / 8; ++c8) {
            v8h x = src[c8];
            #pragma unroll
            for (int j = 0; j < 8; ++j) { float t = (float)x[j]; ss += t * t; }
        }
        float inv = 1.f / fmaxf(sqrtf(ss), 1e-12f);
        #pragma unroll
        for (int c8 = 0; c8 < C / 8; ++c8) {
            v8h x = src[c8], o;
            #pragma unroll
            for (int j = 0; j < 8; ++j) o[j] = (_Float16)((float)x[j] * inv);
            dst[c8] = o;
        }
    } else {
        v8h z = {};
        #pragma unroll
        for (int c8 = 0; c8 < C / 8; ++c8) dst[c8] = z;
    }
}

// ---------------------------------------------------------------------------
// f1 (B,48,64,128): normalized copy -> fn; raw copy -> xcat ch 81..208;
// zero xcat ch 209..255
// ---------------------------------------------------------------------------
__global__ void f1_prep_kernel(const _Float16* __restrict__ f, _Float16* __restrict__ fn,
                               _Float16* __restrict__ x, int Bn) {
    const int C = 128, CX = 256, COFF = 81;
    long i = blockIdx.x * (long)blockDim.x + threadIdx.x;
    long total = (long)Bn * 48 * 64;
    if (i >= total) return;
    const v8h* src = (const v8h*)(f + i * C);
    v8h*       dn  = (v8h*)(fn + i * C);
    _Float16*  xb  = x + i * CX;
    float ss = 0.f;
    #pragma unroll
    for (int c8 = 0; c8 < C / 8; ++c8) {
        v8h v = src[c8];
        #pragma unroll
        for (int j = 0; j < 8; ++j) { float t = (float)v[j]; ss += t * t; }
    }
    float inv = 1.f / fmaxf(sqrtf(ss), 1e-12f);
    #pragma unroll
    for (int c8 = 0; c8 < C / 8; ++c8) {
        v8h v = src[c8], o;
        #pragma unroll
        for (int j = 0; j < 8; ++j) {
            o[j] = (_Float16)((float)v[j] * inv);
            xb[COFF + c8 * 8 + j] = v[j];     // unaligned offset: scalar stores
        }
        dn[c8] = o;
    }
    for (int c = COFF + C; c < CX; ++c) xb[c] = (_Float16)0.f;
}

// ---------------------------------------------------------------------------
// Correlation: xcat[..., d] = sum_c f0p[b, h+dy, w+dx, c] * f1n[b, h, w, c]
// Vectorized b128 channel loads, 4 independent accumulators.
// ---------------------------------------------------------------------------
__global__ void corr_kernel(const _Float16* __restrict__ f0p, const _Float16* __restrict__ f1n,
                            _Float16* __restrict__ x, int Bn) {
    const int C = 128, H = 48, W = 64, ND = 81, Hp = 56, Wp = 72, CX = 256;
    long i = blockIdx.x * (long)blockDim.x + threadIdx.x;
    long total = (long)Bn * ND * H * W;
    if (i >= total) return;
    int b  = (int)(i / ((long)ND * H * W));
    long r = i - (long)b * ND * H * W;
    int d  = (int)(r / (H * W));
    int r2 = (int)(r - (long)d * H * W);
    int h = r2 / W, w = r2 - (r2 / W) * W;
    int dy = d / 9, dx = d - dy * 9;
    const v8h* p0 = (const v8h*)(f0p + (((long)b * Hp + h + dy) * Wp + (w + dx)) * C);
    const v8h* p1 = (const v8h*)(f1n + (((long)b * H + h) * W + w) * C);
    float s0 = 0.f, s1 = 0.f, s2 = 0.f, s3 = 0.f;
    #pragma unroll
    for (int c8 = 0; c8 < C / 8; ++c8) {
        v8h a = p0[c8], bb = p1[c8];
        s0 += (float)a[0] * (float)bb[0] + (float)a[1] * (float)bb[1];
        s1 += (float)a[2] * (float)bb[2] + (float)a[3] * (float)bb[3];
        s2 += (float)a[4] * (float)bb[4] + (float)a[5] * (float)bb[5];
        s3 += (float)a[6] * (float)bb[6] + (float)a[7] * (float)bb[7];
    }
    x[(((long)b * H + h) * W + w) * CX + d] = (_Float16)((s0 + s1) + (s2 + s3));
}

// ---------------------------------------------------------------------------
// Mean pool (B,24,32,256) f16 NHWC -> (B,256) f32. Block per batch, thread=ch.
// ---------------------------------------------------------------------------
__global__ void meanpool_kernel(const _Float16* __restrict__ y, float* __restrict__ out,
                                int Bn) {
    int b = blockIdx.x;
    int c = threadIdx.x;           // 256 channels
    float s = 0.f;
    for (int p = 0; p < 768; ++p) s += (float)y[((long)b * 768 + p) * 256 + c];
    out[b * 256 + c] = s * (1.f / 768.f);
}

// ---------------------------------------------------------------------------
static inline void* bump(char*& p, size_t bytes) {
    void* r = (void*)p;
    p += (bytes + 255) & ~(size_t)255;
    return r;
}
static inline int cdiv(long a, long b) { return (int)((a + b - 1) / b); }

extern "C" void kernel_launch(void* const* d_in, const int* in_sizes, int n_in,
                              void* d_out, int out_size, void* d_ws, size_t ws_size,
                              hipStream_t stream) {
    (void)in_sizes; (void)n_in; (void)out_size; (void)ws_size;
    const int B = 32;

    const float* img0 = (const float*)d_in[0];
    const float* img1 = (const float*)d_in[1];
    const float* W1[6]; const float* W2[6]; const float* W3[6];
    const float* Wf1[6]; const float* Wf2[6];
    for (int j = 0; j < 6; ++j) {
        W1[j]  = (const float*)d_in[2 + j];
        W2[j]  = (const float*)d_in[8 + j];
        W3[j]  = (const float*)d_in[14 + j];
        Wf1[j] = (const float*)d_in[20 + j];
        Wf2[j] = (const float*)d_in[26 + j];
    }

    // ---------------- workspace (all activations NHWC f16) ----------------
    char* ws = (char*)d_ws;
    _Float16* imgh = (_Float16*)bump(ws, (size_t)B * 384 * 512 * 4 * 2);   // reused
    _Float16* a1   = (_Float16*)bump(ws, (size_t)B * 192 * 256 * 32 * 2);
    _Float16* a2   = (_Float16*)bump(ws, (size_t)B * 96 * 128 * 64 * 2);
    _Float16* f0   = (_Float16*)bump(ws, (size_t)B * 48 * 64 * 128 * 2);
    _Float16* f1   = (_Float16*)bump(ws, (size_t)B * 48 * 64 * 128 * 2);
    _Float16* f0p  = (_Float16*)bump(ws, (size_t)B * 56 * 72 * 128 * 2);
    _Float16* f1n  = (_Float16*)bump(ws, (size_t)B * 48 * 64 * 128 * 2);
    _Float16* xcat = (_Float16*)bump(ws, (size_t)B * 48 * 64 * 256 * 2);
    _Float16* y1   = (_Float16*)bump(ws, (size_t)B * 24 * 32 * 256 * 2);
    _Float16* y2   = (_Float16*)bump(ws, (size_t)B * 24 * 32 * 256 * 2);
    _Float16* w1h  = (_Float16*)bump(ws, (size_t)2  * 7  * 512 * 2);   // (Cout/16)*NKS*512
    _Float16* w2h  = (_Float16*)bump(ws, (size_t)4  * 25 * 512 * 2);
    _Float16* w3h  = (_Float16*)bump(ws, (size_t)8  * 18 * 512 * 2);
    _Float16* wf1h = (_Float16*)bump(ws, (size_t)16 * 72 * 512 * 2);
    _Float16* wf2h = (_Float16*)bump(ws, (size_t)16 * 72 * 512 * 2);
    float* bias1  = (float*)bump(ws, 32 * 4);
    float* bias2  = (float*)bump(ws, 64 * 4);
    float* bias3  = (float*)bump(ws, 128 * 4);
    float* biasf1 = (float*)bump(ws, 256 * 4);
    float* biasf2 = (float*)bump(ws, 256 * 4);

    // ---------------- fold BN + shuffle weights into fragment order --------
    auto fold = [&](const float* const* P, _Float16* wo, float* bo, int Co, int Ci,
                    int KH, int KW, int L2C, int KROW, int NKS) {
        int total = (Co / 16) * NKS * 512;
        fold_shuffle_kernel<<<cdiv(total, 256), 256, 0, stream>>>(
            P[0], P[1], P[2], P[3], P[4], P[5], wo, bo, Co, Ci, KH, KW, L2C, KROW, NKS);
    };
    fold(W1,  w1h,  bias1,  32,  3,   7, 7, 2, 32,  7);
    fold(W2,  w2h,  bias2,  64,  32,  5, 5, 5, 160, 25);
    fold(W3,  w3h,  bias3,  128, 64,  3, 3, 6, 192, 18);
    fold(Wf1, wf1h, biasf1, 256, 209, 3, 3, 8, 768, 72);
    fold(Wf2, wf2h, biasf2, 256, 256, 3, 3, 8, 768, 72);

    long npix1 = (long)B * 384 * 512;

    // ---------------- backbone(img0) -> f0 ----------------
    img_to_nhwc4_kernel<<<cdiv(npix1, 256), 256, 0, stream>>>(img0, imgh, B, 384, 512);
    conv_wmma_kernel<4, 2, 384, 512, 32, 192, 256, 7, 7, 2, 3, 2>
        <<<dim3(cdiv((long)B * 192 * 256, 128), 1), 256, 0, stream>>>(imgh, w1h, bias1, a1, B);
    conv_wmma_kernel<32, 5, 192, 256, 64, 96, 128, 5, 5, 2, 2, 4>
        <<<dim3(cdiv((long)B * 96 * 128, 128), 1), 256, 0, stream>>>(a1, w2h, bias2, a2, B);
    conv_wmma_kernel<64, 6, 96, 128, 128, 48, 64, 3, 3, 2, 1, 4>
        <<<dim3(cdiv((long)B * 48 * 64, 128), 2), 256, 0, stream>>>(a2, w3h, bias3, f0, B);

    // ---------------- backbone(img1) -> f1 ----------------
    img_to_nhwc4_kernel<<<cdiv(npix1, 256), 256, 0, stream>>>(img1, imgh, B, 384, 512);
    conv_wmma_kernel<4, 2, 384, 512, 32, 192, 256, 7, 7, 2, 3, 2>
        <<<dim3(cdiv((long)B * 192 * 256, 128), 1), 256, 0, stream>>>(imgh, w1h, bias1, a1, B);
    conv_wmma_kernel<32, 5, 192, 256, 64, 96, 128, 5, 5, 2, 2, 4>
        <<<dim3(cdiv((long)B * 96 * 128, 128), 1), 256, 0, stream>>>(a1, w2h, bias2, a2, B);
    conv_wmma_kernel<64, 6, 96, 128, 128, 48, 64, 3, 3, 2, 1, 4>
        <<<dim3(cdiv((long)B * 48 * 64, 128), 2), 256, 0, stream>>>(a2, w3h, bias3, f1, B);

    // ---------------- correlation + concat ----------------
    l2norm_pad_kernel<<<cdiv((long)B * 56 * 72, 256), 256, 0, stream>>>(f0, f0p, B);
    f1_prep_kernel<<<cdiv((long)B * 48 * 64, 256), 256, 0, stream>>>(f1, f1n, xcat, B);
    corr_kernel<<<cdiv((long)B * 81 * 48 * 64, 256), 256, 0, stream>>>(f0p, f1n, xcat, B);

    // ---------------- fusion convs + pool ----------------
    conv_wmma_kernel<256, 8, 48, 64, 256, 24, 32, 3, 3, 2, 1, 4>
        <<<dim3(cdiv((long)B * 24 * 32, 128), 4), 256, 0, stream>>>(xcat, wf1h, biasf1, y1, B);
    conv_wmma_kernel<256, 8, 24, 32, 256, 24, 32, 3, 3, 1, 1, 4>
        <<<dim3(cdiv((long)B * 24 * 32, 128), 4), 256, 0, stream>>>(y1, wf2h, biasf2, y2, B);
    meanpool_kernel<<<B, 256, 0, stream>>>(y2, (float*)d_out, B);
}
